// ChebNet_38397007626279
// MI455X (gfx1250) — matrix operand order repaired and checked by reading, not compiled, not run on hardware
//
#include <hip/hip_runtime.h>

typedef __attribute__((ext_vector_type(16))) _Float16 v16h;
typedef __attribute__((ext_vector_type(8)))  float    v8f;

#define GG 512   // graphs
#define CC 10    // classes

// Precomputed B-fragment buffer geometry:
// 72 Cheb fragments: [layer(3)][w(3)][k0t(2)][tile(4)]  +  8 lin1 fragments: [k0t(2)][tile(4)]
// each fragment: [lane(32)][16 x f16]  (32 bytes per lane, lanes contiguous)
#define NFRAG_CHEB 72
#define NFRAG_TOT  80

// ---------------- elementwise / edge kernels ----------------

__global__ void fill_zero(float* __restrict__ p, int n) {
  int i = blockIdx.x * blockDim.x + threadIdx.x;
  if (i < n) p[i] = 0.0f;
}

__global__ void neg_copy(float* __restrict__ d, const float* __restrict__ s, int n) {
  int i = blockIdx.x * blockDim.x + threadIdx.x;
  if (i < n) d[i] = -s[i];
}

__global__ void deg_scatter(const int* __restrict__ src, const int* __restrict__ dst,
                            const float* __restrict__ ew, float* __restrict__ deg, int E) {
  int e = blockIdx.x * blockDim.x + threadIdx.x;
  if (e >= E) return;
  int s = src[e];
  float w = (s == dst[e]) ? 0.0f : ew[e];
  if (w != 0.0f) atomicAdd(&deg[s], w);
}

__global__ void make_dis(float* __restrict__ deg, int n) {
  int i = blockIdx.x * blockDim.x + threadIdx.x;
  if (i >= n) return;
  float d = deg[i];
  deg[i] = (d > 0.0f) ? rsqrtf(d) : 0.0f;  // in-place deg -> D^{-1/2}
}

__global__ void make_norm(const int* __restrict__ src, const int* __restrict__ dst,
                          const float* __restrict__ ew, const float* __restrict__ dis,
                          float* __restrict__ norm, int E) {
  int e = blockIdx.x * blockDim.x + threadIdx.x;
  if (e >= E) return;
  int s = src[e], d = dst[e];
  float w = (s == d) ? 0.0f : ew[e];
  norm[e] = -dis[s] * w * dis[d];
}

// out[dst] += scale * norm * h[src]   (one thread per (edge, 4-feature group))
__global__ void prop_scatter(const float* __restrict__ h, const int* __restrict__ src,
                             const int* __restrict__ dst, const float* __restrict__ norm,
                             float* __restrict__ out, float scale, int E) {
  int idx = blockIdx.x * blockDim.x + threadIdx.x;
  if (idx >= E * 16) return;
  int e = idx >> 4;
  int f = (idx & 15) << 2;
  float nw = norm[e] * scale;
  if (nw == 0.0f) return;
  const float4 v = *(const float4*)(h + (size_t)src[e] * 64 + f);
  float* o = out + (size_t)dst[e] * 64 + f;
  atomicAdd(o + 0, nw * v.x);
  atomicAdd(o + 1, nw * v.y);
  atomicAdd(o + 2, nw * v.z);
  atomicAdd(o + 3, nw * v.w);
}

// ---------------- B-fragment precompute (f32 weights -> f16 WMMA layout) ----------------
// ISA B layout (16-bit, 32x16): lanes 0-15 hold K=k0+{0..7,16..23}, lanes 16-31 K=k0+{8..15,24..31},
// col = n0 + (lane&15); element e of the v16h is K=kb+e (e<8) or kb+e+8 (e>=8).

__global__ void build_wfrags(const float* __restrict__ W1, const float* __restrict__ W2,
                             const float* __restrict__ W3, const float* __restrict__ l1w,
                             _Float16* __restrict__ out) {
  int t = blockIdx.x * blockDim.x + threadIdx.x;
  if (t >= NFRAG_TOT * 32) return;
  int lane = t & 31;
  int frag = t >> 5;  // 0..79
  const float* Wsrc;
  int k0, n0;
  if (frag < NFRAG_CHEB) {
    int layer = frag / 24;
    int rem   = frag % 24;       // w*8 + k0t*4 + tile
    int w     = rem >> 3;
    int k0t   = (rem >> 2) & 1;
    int tile  = rem & 3;
    const float* Wl = (layer == 0) ? W1 : (layer == 1) ? W2 : W3;
    Wsrc = Wl + w * 64 * 64;
    k0 = k0t * 32;
    n0 = tile * 16;
  } else {
    int rem  = frag - NFRAG_CHEB;  // k0t*4 + tile
    Wsrc = l1w;
    k0 = (rem >> 2) * 32;
    n0 = (rem & 3) * 16;
  }
  int kb  = k0 + ((lane & 16) ? 8 : 0);
  int col = n0 + (lane & 15);
  _Float16* o = out + (size_t)t * 16;
#pragma unroll
  for (int e = 0; e < 16; ++e) o[e] = (_Float16)Wsrc[(size_t)(kb + e + ((e & 8) ? 8 : 0)) * 64 + col];
}

// ---------------- WMMA A-fragment (f32 mem -> f16 regs, vectorized) ----------------
// A layout (16-bit, 16x32): lane<16: row=lane,    K = k0+{0..7, 16..23}
//                           lane>=16: row=lane-16, K = k0+{8..15, 24..31}
__device__ __forceinline__ v16h load_a_frag(const float* __restrict__ A, int row, int k0, int lane) {
  const float* p = A + (size_t)row * 64 + k0 + ((lane & 16) ? 8 : 0);
  float4 x0 = *(const float4*)(p);
  float4 x1 = *(const float4*)(p + 4);
  float4 x2 = *(const float4*)(p + 16);
  float4 x3 = *(const float4*)(p + 20);
  v16h a;
  a[0]  = (_Float16)x0.x; a[1]  = (_Float16)x0.y; a[2]  = (_Float16)x0.z; a[3]  = (_Float16)x0.w;
  a[4]  = (_Float16)x1.x; a[5]  = (_Float16)x1.y; a[6]  = (_Float16)x1.z; a[7]  = (_Float16)x1.w;
  a[8]  = (_Float16)x2.x; a[9]  = (_Float16)x2.y; a[10] = (_Float16)x2.z; a[11] = (_Float16)x2.w;
  a[12] = (_Float16)x3.x; a[13] = (_Float16)x3.y; a[14] = (_Float16)x3.z; a[15] = (_Float16)x3.w;
  return a;
}

// ---------------- fused Chebyshev GEMM: out = relu(Tx0@W0 + Tx1@W1 + Tx2@W2 + b) ----------------
// One wave computes a 16-row x 64-col tile; f32 accumulation; 24 WMMAs per wave.
// Bf points at this layer's 24 precomputed fragments.

__global__ void cheb_gemm_fused(const float* __restrict__ Tx0, const float* __restrict__ Tx1,
                                const float* __restrict__ Tx2, const v16h* __restrict__ Bf,
                                const float* __restrict__ bias, float* __restrict__ out,
                                int nrows) {
  int wave = (blockIdx.x * blockDim.x + threadIdx.x) >> 5;
  int lane = threadIdx.x & 31;
  int row0 = wave * 16;
  if (row0 >= nrows) return;           // wave-uniform exit: EXEC stays all-ones
  int arow = row0 + (lane & 15);
  if (arow >= nrows) arow = nrows - 1; // clamp scalar loads only
  v8f acc0 = {}, acc1 = {}, acc2 = {}, acc3 = {};
  const float* mats[3] = {Tx0, Tx1, Tx2};
#pragma unroll
  for (int w = 0; w < 3; ++w) {
    const float* A = mats[w];
#pragma unroll
    for (int k0t = 0; k0t < 2; ++k0t) {
      v16h a = load_a_frag(A, arow, k0t * 32, lane);
      const v16h* bp = Bf + (size_t)(w * 8 + k0t * 4) * 32 + lane;
      acc0 = __builtin_amdgcn_wmma_f32_16x16x32_f16(false, a, false, bp[0],  (short)0, acc0, false, false);
      acc1 = __builtin_amdgcn_wmma_f32_16x16x32_f16(false, a, false, bp[32], (short)0, acc1, false, false);
      acc2 = __builtin_amdgcn_wmma_f32_16x16x32_f16(false, a, false, bp[64], (short)0, acc2, false, false);
      acc3 = __builtin_amdgcn_wmma_f32_16x16x32_f16(false, a, false, bp[96], (short)0, acc3, false, false);
    }
  }
  int mb = row0 + ((lane & 16) ? 8 : 0);
  int cb = lane & 15;
  float bi0 = bias[cb], bi1 = bias[16 + cb], bi2 = bias[32 + cb], bi3 = bias[48 + cb];
#pragma unroll
  for (int g = 0; g < 8; ++g) {
    int m = mb + g;
    if (m < nrows) {
      float* orow = out + (size_t)m * 64 + cb;
      orow[0]  = fmaxf(acc0[g] + bi0, 0.0f);
      orow[16] = fmaxf(acc1[g] + bi1, 0.0f);
      orow[32] = fmaxf(acc2[g] + bi2, 0.0f);
      orow[48] = fmaxf(acc3[g] + bi3, 0.0f);
    }
  }
}

// out = relu(A@W + b), A nrows x 64 (used for lin1); Bf = 8 precomputed fragments
__global__ void gemm_relu(const float* __restrict__ A, const v16h* __restrict__ Bf,
                          const float* __restrict__ bias, float* __restrict__ out, int nrows) {
  int wave = (blockIdx.x * blockDim.x + threadIdx.x) >> 5;
  int lane = threadIdx.x & 31;
  int row0 = wave * 16;
  if (row0 >= nrows) return;
  int arow = row0 + (lane & 15);
  if (arow >= nrows) arow = nrows - 1;
  v8f acc0 = {}, acc1 = {}, acc2 = {}, acc3 = {};
#pragma unroll
  for (int k0t = 0; k0t < 2; ++k0t) {
    v16h a = load_a_frag(A, arow, k0t * 32, lane);
    const v16h* bp = Bf + (size_t)(k0t * 4) * 32 + lane;
    acc0 = __builtin_amdgcn_wmma_f32_16x16x32_f16(false, a, false, bp[0],  (short)0, acc0, false, false);
    acc1 = __builtin_amdgcn_wmma_f32_16x16x32_f16(false, a, false, bp[32], (short)0, acc1, false, false);
    acc2 = __builtin_amdgcn_wmma_f32_16x16x32_f16(false, a, false, bp[64], (short)0, acc2, false, false);
    acc3 = __builtin_amdgcn_wmma_f32_16x16x32_f16(false, a, false, bp[96], (short)0, acc3, false, false);
  }
  int mb = row0 + ((lane & 16) ? 8 : 0);
  int cb = lane & 15;
  float bi0 = bias[cb], bi1 = bias[16 + cb], bi2 = bias[32 + cb], bi3 = bias[48 + cb];
#pragma unroll
  for (int g = 0; g < 8; ++g) {
    int m = mb + g;
    if (m < nrows) {
      float* orow = out + (size_t)m * 64 + cb;
      orow[0]  = fmaxf(acc0[g] + bi0, 0.0f);
      orow[16] = fmaxf(acc1[g] + bi1, 0.0f);
      orow[32] = fmaxf(acc2[g] + bi2, 0.0f);
      orow[48] = fmaxf(acc3[g] + bi3, 0.0f);
    }
  }
}

// ---------------- pooling + head ----------------

__global__ void pool_scatter(const float* __restrict__ h, const int* __restrict__ batch,
                             float* __restrict__ pool, float* __restrict__ cnt, int n) {
  int idx = blockIdx.x * blockDim.x + threadIdx.x;
  if (idx >= n * 64) return;
  int node = idx >> 6, f = idx & 63;
  int b = batch[node];
  atomicAdd(&pool[(size_t)b * 64 + f], h[idx]);
  if (f == 0) atomicAdd(&cnt[b], 1.0f);
}

__global__ void pool_div(const float* __restrict__ pool, const float* __restrict__ cnt,
                         float* __restrict__ g) {
  int i = blockIdx.x * blockDim.x + threadIdx.x;
  if (i >= GG * 64) return;
  g[i] = pool[i] / fmaxf(cnt[i >> 6], 1.0f);
}

__global__ void head_kernel(const float* __restrict__ g1, const float* __restrict__ w,
                            const float* __restrict__ b, float* __restrict__ out, int Gn) {
  int g = blockIdx.x * blockDim.x + threadIdx.x;
  if (g >= Gn) return;
  float z[CC];
#pragma unroll
  for (int c = 0; c < CC; ++c) z[c] = b[c];
  for (int f = 0; f < 64; ++f) {
    float v = g1[(size_t)g * 64 + f];
#pragma unroll
    for (int c = 0; c < CC; ++c) z[c] += v * w[f * CC + c];
  }
  float m = z[0];
#pragma unroll
  for (int c = 1; c < CC; ++c) m = fmaxf(m, z[c]);
  float s = 0.0f;
#pragma unroll
  for (int c = 0; c < CC; ++c) s += expf(z[c] - m);
  float lse = m + logf(s);
#pragma unroll
  for (int c = 0; c < CC; ++c) out[(size_t)g * CC + c] = z[c] - lse;
}

// ---------------- launch ----------------

static inline int ceil_div(int a, int b) { return (a + b - 1) / b; }

extern "C" void kernel_launch(void* const* d_in, const int* in_sizes, int n_in,
                              void* d_out, int out_size, void* d_ws, size_t ws_size,
                              hipStream_t stream) {
  (void)n_in; (void)out_size; (void)ws_size;
  const float* x    = (const float*)d_in[0];
  const int*   ei   = (const int*)d_in[1];
  const float* ew   = (const float*)d_in[2];
  const int*   bat  = (const int*)d_in[3];
  const float* W1   = (const float*)d_in[4];
  const float* b1   = (const float*)d_in[5];
  const float* W2   = (const float*)d_in[6];
  const float* b2   = (const float*)d_in[7];
  const float* W3   = (const float*)d_in[8];
  const float* b3   = (const float*)d_in[9];
  const float* l1w  = (const float*)d_in[10];
  const float* l1b  = (const float*)d_in[11];
  const float* l2w  = (const float*)d_in[12];
  const float* l2b  = (const float*)d_in[13];

  const int N = in_sizes[0] / 64;
  const int E = in_sizes[2];
  const int* src = ei;       // edge_index[0]
  const int* dst = ei + E;   // edge_index[1]

  // workspace: fragment buffer first (32B-aligned from base), then float buffers
  _Float16* wfrag = (_Float16*)d_ws;                       // 80*32*16 f16 = 80 KiB
  float* p    = (float*)((char*)d_ws + (size_t)NFRAG_TOT * 32 * 16 * sizeof(_Float16));
  float* norm = p; p += E;
  float* dis  = p; p += N;
  float* Tx1  = p; p += (size_t)N * 64;
  float* Tx2  = p; p += (size_t)N * 64;
  float* hA   = p; p += (size_t)N * 64;
  float* hB   = p; p += (size_t)N * 64;
  float* pool = p; p += GG * 64;
  float* cnt  = p; p += GG;
  float* gbuf = p; p += GG * 64;
  float* g1   = p; p += GG * 64;

  const int B = 256;

  // Precompute all B fragments (f16, WMMA register layout)
  build_wfrags<<<ceil_div(NFRAG_TOT * 32, B), B, 0, stream>>>(W1, W2, W3, l1w, wfrag);

  // Edge normalization: deg -> D^{-1/2} -> norm = -dis[src]*w*dis[dst]
  fill_zero<<<ceil_div(N, B), B, 0, stream>>>(dis, N);
  deg_scatter<<<ceil_div(E, B), B, 0, stream>>>(src, dst, ew, dis, E);
  make_dis<<<ceil_div(N, B), B, 0, stream>>>(dis, N);
  make_norm<<<ceil_div(E, B), B, 0, stream>>>(src, dst, ew, dis, norm, E);

  const int gemm_blocks = ceil_div(ceil_div(N, 16) * 32, B);
  const int propT = E * 16;

  const float* in = x;
  const float* bs[3]   = {b1, b2, b3};
  float*       outs[3] = {hA, hB, hA};
  for (int l = 0; l < 3; ++l) {
    // Tx1 = L_hat @ in
    fill_zero<<<ceil_div(N * 64, B), B, 0, stream>>>(Tx1, N * 64);
    prop_scatter<<<ceil_div(propT, B), B, 0, stream>>>(in, src, dst, norm, Tx1, 1.0f, E);
    // Tx2 = 2 * L_hat @ Tx1 - in   (init with -in, scatter scaled by 2)
    neg_copy<<<ceil_div(N * 64, B), B, 0, stream>>>(Tx2, in, N * 64);
    prop_scatter<<<ceil_div(propT, B), B, 0, stream>>>(Tx1, src, dst, norm, Tx2, 2.0f, E);
    // out = relu(in@W0 + Tx1@W1 + Tx2@W2 + b)
    const v16h* Bf = (const v16h*)(wfrag) + (size_t)l * 24 * 32;
    cheb_gemm_fused<<<gemm_blocks, B, 0, stream>>>(in, Tx1, Tx2, Bf, bs[l], outs[l], N);
    in = outs[l];
  }

  // global mean pool -> lin1(relu) -> lin2 + log_softmax
  fill_zero<<<ceil_div(GG * 64 + GG, B), B, 0, stream>>>(pool, GG * 64 + GG);
  pool_scatter<<<ceil_div(N * 64, B), B, 0, stream>>>(hA, bat, pool, cnt, N);
  pool_div<<<ceil_div(GG * 64, B), B, 0, stream>>>(pool, cnt, gbuf);
  const v16h* Bf1 = (const v16h*)(wfrag) + (size_t)NFRAG_CHEB * 32;
  gemm_relu<<<ceil_div(ceil_div(GG, 16) * 32, B), B, 0, stream>>>(gbuf, Bf1, l1b, g1, GG);
  head_kernel<<<ceil_div(GG, B), B, 0, stream>>>(g1, l2w, l2b, (float*)d_out, GG);
}
